// SimpleRNN_32323923869937
// MI455X (gfx1250) — matrix operand order repaired and checked by reading, not compiled, not run on hardware
//
#include <hip/hip_runtime.h>

// Problem sizes (fixed by the reference).
#define SEQ   512
#define NB    64      // batch
#define DH    1024    // hidden
#define DIN   1024    // input feature

typedef __attribute__((ext_vector_type(16))) __bf16 v16bf;
typedef __attribute__((ext_vector_type(8)))  float  v8f;

// ---------------------------------------------------------------------------
// CDNA5 helpers
// ---------------------------------------------------------------------------
__device__ __forceinline__ v8f wmma_bf16(v16bf a, v16bf b, v8f c) {
  // 8 args: (neg_a, A, neg_b, B, c_mod, C, reuse_a, reuse_b)
  return __builtin_amdgcn_wmma_f32_16x16x32_bf16(
      false, a, false, b, (short)0, c, false, false);
}

// Low 32 bits of a generic pointer to LDS == LDS byte offset (ISA 10.2).
__device__ __forceinline__ unsigned lds_off(const void* p) {
  return (unsigned)(unsigned long long)(uintptr_t)p;
}

// Async DMA: 16 bytes per active lane, global -> LDS, tracked by ASYNCcnt.
__device__ __forceinline__ void async_b128(unsigned lds_dst, const void* gsrc) {
  asm volatile("global_load_async_to_lds_b128 %0, %1, off"
               :: "v"(lds_dst), "v"(gsrc) : "memory");
}
__device__ __forceinline__ void wait_async0() {
  asm volatile("s_wait_asynccnt 0" ::: "memory");
}

// A-matrix fragment (16x32 MxK slice at column k0), per ISA 7.12.2:
//   lanes 0-15: M=lane, VGPR0..3 -> K=0..7, VGPR4..7 -> K=16..23
//   lanes16-31: M=lane-16,         K=8..15,            K=24..31
__device__ __forceinline__ v16bf frag_a(const __bf16* tile, int lda, int k0, int lane) {
  const int m    = lane & 15;
  const int half = (lane >> 4) & 1;
  v16bf a;
#pragma unroll
  for (int v = 0; v < 8; ++v) {
    const int k = k0 + ((v < 4) ? (2 * v) : (16 + 2 * (v - 4))) + 8 * half;
    a[2 * v]     = tile[m * lda + k];
    a[2 * v + 1] = tile[m * lda + k + 1];
  }
  return a;
}

// B-matrix fragment (32x16 KxN slice at k0); tile holds B^T rows ([n][k]).
//   lanes 0-15: N=lane, K=0..7,16..23; lanes16-31: N=lane-16, K=8..15,24..31
__device__ __forceinline__ v16bf frag_b(const __bf16* tile, int ldb, int k0, int lane) {
  const int n    = lane & 15;
  const int half = (lane >> 4) & 1;
  v16bf b;
#pragma unroll
  for (int v = 0; v < 8; ++v) {
    const int k = k0 + ((v < 4) ? (2 * v) : (16 + 2 * (v - 4))) + 8 * half;
    b[2 * v]     = tile[n * ldb + k];
    b[2 * v + 1] = tile[n * ldb + k + 1];
  }
  return b;
}

// ---------------------------------------------------------------------------
// Phase 0: one-time f32 -> bf16 conversion of W_hh and h0 into workspace.
// ---------------------------------------------------------------------------
__global__ void convert_kernel(const float* __restrict__ whh_f,
                               const float* __restrict__ h0_f,
                               __bf16* __restrict__ whh_b,
                               __bf16* __restrict__ h0_b) {
  const int i = blockIdx.x * blockDim.x + threadIdx.x;
  if (i < DH * DH) whh_b[i] = (__bf16)whh_f[i];
  if (i < NB * DH) h0_b[i] = (__bf16)h0_f[i];
}

// ---------------------------------------------------------------------------
// Phase 1: x_proj = x @ W_ih^T + b_ih + b_hh   (M=SEQ*NB=32768, N=DH, K=DIN)
// HBM-bound (~270 MB @ 23.3 TB/s). 256 threads / 8 waves, block tile 64Mx128N,
// wave tile 32x32 (2x2 WMMA), K-step 32, float4 staging (v_cvt_pk_bf16_f32).
// ---------------------------------------------------------------------------
#define LDA_PAD 34   // 17-dword row stride -> conflict-free gathers

__global__ void xproj_kernel(const float* __restrict__ x,
                             const float* __restrict__ Wih,
                             const float* __restrict__ bih,
                             const float* __restrict__ bhh,
                             float* __restrict__ out) {
  __shared__ __bf16 As[64 * LDA_PAD];    // 64 rows(M)  x 32 K
  __shared__ __bf16 Bs[128 * LDA_PAD];   // 128 rows(N) x 32 K (W row-major = B^T)

  const int m0g  = blockIdx.y * 64;
  const int n0g  = blockIdx.x * 128;
  const int tid  = threadIdx.x;
  const int wid  = tid >> 5;
  const int lane = tid & 31;
  const int wm   = (wid >> 2) * 32;   // 0 / 32
  const int wn   = (wid & 3) * 32;    // 0 / 32 / 64 / 96

  v8f acc[2][2];
  acc[0][0] = {}; acc[0][1] = {}; acc[1][0] = {}; acc[1][1] = {};

  for (int k0 = 0; k0 < DIN; k0 += 32) {
    // A tile: 512 float4 chunks, exactly 2 per thread (no predication).
#pragma unroll
    for (int u = 0; u < 2; ++u) {
      const int c = tid + u * 256;
      const int i = c >> 3, j = c & 7;
      const float4 v4 = *(const float4*)&x[(size_t)(m0g + i) * DIN + k0 + j * 4];
      __bf16* d = &As[i * LDA_PAD + j * 4];
      d[0] = (__bf16)v4.x; d[1] = (__bf16)v4.y; d[2] = (__bf16)v4.z; d[3] = (__bf16)v4.w;
    }
    // B tile: 1024 float4 chunks, exactly 4 per thread.
#pragma unroll
    for (int u = 0; u < 4; ++u) {
      const int c = tid + u * 256;
      const int i = c >> 3, j = c & 7;
      const float4 v4 = *(const float4*)&Wih[(size_t)(n0g + i) * DIN + k0 + j * 4];
      __bf16* d = &Bs[i * LDA_PAD + j * 4];
      d[0] = (__bf16)v4.x; d[1] = (__bf16)v4.y; d[2] = (__bf16)v4.z; d[3] = (__bf16)v4.w;
    }
    __syncthreads();

    const v16bf a0 = frag_a(As + (wm + 0)  * LDA_PAD, LDA_PAD, 0, lane);
    const v16bf a1 = frag_a(As + (wm + 16) * LDA_PAD, LDA_PAD, 0, lane);
    const v16bf b0 = frag_b(Bs + (wn + 0)  * LDA_PAD, LDA_PAD, 0, lane);
    const v16bf b1 = frag_b(Bs + (wn + 16) * LDA_PAD, LDA_PAD, 0, lane);

    acc[0][0] = wmma_bf16(a0, b0, acc[0][0]);
    acc[0][1] = wmma_bf16(a0, b1, acc[0][1]);
    acc[1][0] = wmma_bf16(a1, b0, acc[1][0]);
    acc[1][1] = wmma_bf16(a1, b1, acc[1][1]);
    __syncthreads();
  }

  // Epilogue: C/D layout lane -> N, vgpr r -> M = r + 8*(lane>=16).
  const int half = lane >> 4;
#pragma unroll
  for (int mi = 0; mi < 2; ++mi) {
#pragma unroll
    for (int ni = 0; ni < 2; ++ni) {
      const int n = n0g + wn + ni * 16 + (lane & 15);
      const float bias = bih[n] + bhh[n];
#pragma unroll
      for (int r = 0; r < 8; ++r) {
        const int m = m0g + wm + mi * 16 + r + 8 * half;
        out[(size_t)m * DH + n] = acc[mi][ni][r] + bias;
      }
    }
  }
}

// ---------------------------------------------------------------------------
// Phase 2: persistent scan kernel. 64 blocks x 128 threads (4 waves), block b
// owns N columns [16b, 16b+16) for ALL 512 timesteps.
//  - W_hh strip (16x1024 bf16, 33 KB) pinned in LDS once (async DMA).
//  - FULL h_{t-1} (64x1024 bf16, 132 KB) staged per step in ONE async batch
//    -> one s_wait_asynccnt + one barrier per step, all DMA in flight at once.
//  - Per wave: one 16x16 tile, K=1024 -> 32 WMMA/step on 2 accumulators.
//  - Device-wide barrier per step via per-t arrival counters (agent scope).
// 165 KB of the 320 KB CDNA5 LDS -> 1 block per WGP (matches 64-block grid).
// ---------------------------------------------------------------------------
#define AS_STR 1032   // 1024+8: rows 16B-aligned (2064 B), 516 dw = 4 mod 64
#define BS_STR 1032

__global__ __launch_bounds__(128, 1)
void rnn_scan_kernel(float* __restrict__ out,          // [SEQ][NB][DH] xp in / h out
                     const __bf16* __restrict__ h0b,   // [NB][DH]
                     const __bf16* __restrict__ whh,   // [DH][DH] bf16 row-major
                     __bf16* __restrict__ hbuf,        // 2 x [NB][DH] ping-pong
                     unsigned* __restrict__ bar) {     // [SEQ] arrival counters (zeroed)
  __shared__ __align__(16) __bf16 Bs[16 * BS_STR];   // resident W_hh strip (33 KB)
  __shared__ __align__(16) __bf16 As[64 * AS_STR];   // full h_prev        (132 KB)

  const int n0   = blockIdx.x * 16;
  const int tid  = threadIdx.x;
  const int wid  = tid >> 5;
  const int lane = tid & 31;
  const int mb   = wid * 16;           // wave's M (batch) offset
  const unsigned nblk = gridDim.x;

  // Pin W_hh strip: thread tid copies 16B chunk tid of each of 16 rows.
#pragma unroll
  for (int u = 0; u < 16; ++u)
    async_b128(lds_off(&Bs[u * BS_STR + tid * 8]),
               whh + (size_t)(n0 + u) * DH + tid * 8);
  wait_async0();
  __syncthreads();

  for (int t = 0; t < SEQ; ++t) {
    const __bf16* hp = (t == 0) ? h0b : hbuf + (size_t)((t - 1) & 1) * (NB * DH);
    __bf16*       ho = hbuf + (size_t)(t & 1) * (NB * DH);
    float*        y  = out + (size_t)t * NB * DH;

    // Stage the whole h_prev: 64 rows, chunk tid of each row, all in flight.
#pragma unroll
    for (int u = 0; u < 64; ++u)
      async_b128(lds_off(&As[u * AS_STR + tid * 8]),
                 hp + u * DH + tid * 8);
    wait_async0();
    __syncthreads();

    v8f acc0 = {}, acc1 = {};
#pragma unroll
    for (int k0 = 0; k0 < DH; k0 += 64) {
      const v16bf a0 = frag_a(As + mb * AS_STR, AS_STR, k0, lane);
      const v16bf b0 = frag_b(Bs, BS_STR, k0, lane);
      acc0 = wmma_bf16(a0, b0, acc0);
      const v16bf a1 = frag_a(As + mb * AS_STR, AS_STR, k0 + 32, lane);
      const v16bf b1 = frag_b(Bs, BS_STR, k0 + 32, lane);
      acc1 = wmma_bf16(a1, b1, acc1);
    }
    const v8f acc = acc0 + acc1;

    // h_t = tanh(xp_t + acc); overwrite y in place, emit bf16 for next step.
    const int n    = n0 + (lane & 15);
    const int half = lane >> 4;
#pragma unroll
    for (int r = 0; r < 8; ++r) {
      const int m = mb + r + 8 * half;
      const float v = tanhf(acc[r] + y[m * DH + n]);
      y[m * DH + n]  = v;
      ho[m * DH + n] = (__bf16)v;
    }

    // ---- device-wide barrier: release h_t, acquire before next step ----
    // (also orders next step's As DMA after all waves finished reading As)
    __threadfence();
    __syncthreads();
    if (tid == 0) {
      __hip_atomic_fetch_add(&bar[t], 1u, __ATOMIC_RELEASE, __HIP_MEMORY_SCOPE_AGENT);
      while (__hip_atomic_load(&bar[t], __ATOMIC_ACQUIRE, __HIP_MEMORY_SCOPE_AGENT) < nblk)
        __builtin_amdgcn_s_sleep(1);
    }
    __syncthreads();
  }
}

// ---------------------------------------------------------------------------
// Host-side launch
// ---------------------------------------------------------------------------
extern "C" void kernel_launch(void* const* d_in, const int* in_sizes, int n_in,
                              void* d_out, int out_size, void* d_ws, size_t ws_size,
                              hipStream_t stream) {
  const float* x   = (const float*)d_in[0];  // [SEQ, NB, DIN]
  const float* h0  = (const float*)d_in[1];  // [NB, DH]
  const float* Wih = (const float*)d_in[2];  // [DH, DIN]
  const float* bih = (const float*)d_in[3];  // [DH]
  const float* Whh = (const float*)d_in[4];  // [DH, DH]
  const float* bhh = (const float*)d_in[5];  // [DH]
  float* out = (float*)d_out;                // [SEQ,NB,DH] + [NB,DH] h_last

  // ws: W_hh bf16 (2MB) | h0 bf16 (128KB) | hbuf ping+pong (256KB) | bar (2KB)
  char* ws = (char*)d_ws;
  __bf16*  whh_b = (__bf16*)ws;
  __bf16*  h0b   = (__bf16*)(ws + (size_t)DH * DH * sizeof(__bf16));
  __bf16*  hbuf  = h0b + NB * DH;
  unsigned* bar  = (unsigned*)(hbuf + 2 * NB * DH);

  // Phase 0: conversions (W_hh then resident in the 192 MB L2).
  convert_kernel<<<(DH * DH + 255) / 256, 256, 0, stream>>>(Whh, h0, whh_b, h0b);

  // Phase 1: batched input projection into d_out.
  dim3 gridA(DH / 128, (SEQ * NB) / 64);
  xproj_kernel<<<gridA, 256, 0, stream>>>(x, Wih, bih, bhh, out);

  // Phase 2: single persistent kernel runs the whole 512-step recurrence.
  hipMemsetAsync(bar, 0, SEQ * sizeof(unsigned), stream);
  rnn_scan_kernel<<<NB, 128, 0, stream>>>(out, h0b, whh_b, hbuf, bar);

  // h_last = output[SEQ-1] -> tail of d_out.
  hipMemcpyAsync(out + (size_t)SEQ * NB * DH,
                 out + (size_t)(SEQ - 1) * NB * DH,
                 (size_t)NB * DH * sizeof(float),
                 hipMemcpyDeviceToDevice, stream);
}
